// Codebook_39702677684263
// MI455X (gfx1250) — compile-verified, matrix operand order checked by reading, not run on hardware
//
#include <hip/hip_runtime.h>

// ---------------------------------------------------------------------------
// VQ quantize for MI455X (gfx1250, wave32, WMMA + async-to-LDS).
//   argmin_k ||x_m - c_k||^2 == argmin_k (||c_k||^2 - 2 x_m.c_k).
//   Cross term = 16384x4096x512 GEMM via v_wmma_f32_16x16x32_bf16 with an
//   f32 -> bf16 hi/lo split (hi*hi + hi*lo + lo*hi) for near-f32 accuracy.
//   Codebook is pre-converted once to packed bf16 hi/lo (8 MB, L2-resident);
//   the GEMM stages it with global_load_async_to_lds_b128 (ASYNCcnt,
//   double-buffered LDS, one barrier per chunk). Codes tile converted once
//   per block and held in LDS for the whole codebook sweep.
// ---------------------------------------------------------------------------

typedef __attribute__((ext_vector_type(16))) __bf16 v16bf;
typedef __attribute__((ext_vector_type(8)))  float  v8f;

#define DDIM   512
#define KCB    4096
#define MROWS  16384
#define MT     64            // code rows per block (4 waves x 16 rows)
#define NT     64            // codebook rows per tile
#define CH     32            // d-chunk per WMMA step
#define NCH    (DDIM / CH)   // 16
#define NTILES (KCB / NT)    // 64
#define GTOT   (NTILES * NCH)
#define AROWB  2064          // A LDS row stride bytes: 16 chunks*128B + 16 pad
#define BROWB  144           // B LDS row stride bytes: 128B + 16 pad
#define CBROWB 2048          // packed codebook row bytes (16 chunks * [32 hi | 32 lo] bf16)

static __device__ __forceinline__ unsigned short bf16_rne(float x) {
    unsigned u = __builtin_bit_cast(unsigned, x);
    u += 0x7FFFu + ((u >> 16) & 1u);          // round-to-nearest-even
    return (unsigned short)(u >> 16);
}

static __device__ __forceinline__ void cvt_hilo4(const float4& x, ushort4& h, ushort4& l) {
    h.x = bf16_rne(x.x); h.y = bf16_rne(x.y); h.z = bf16_rne(x.z); h.w = bf16_rne(x.w);
    l.x = bf16_rne(x.x - __builtin_bit_cast(float, (unsigned)h.x << 16));
    l.y = bf16_rne(x.y - __builtin_bit_cast(float, (unsigned)h.y << 16));
    l.z = bf16_rne(x.z - __builtin_bit_cast(float, (unsigned)h.z << 16));
    l.w = bf16_rne(x.w - __builtin_bit_cast(float, (unsigned)h.w << 16));
}

// ---- pass 1: codebook -> packed bf16 hi/lo + ||c_k||^2 (fused, one sweep) --
__global__ __launch_bounds__(128) void vq_prep_kernel(
    const float* __restrict__ codebook, unsigned short* __restrict__ cbPk,
    float* __restrict__ cbSq)
{
    const int k = blockIdx.x, tid = threadIdx.x;
    const int d = tid * 4, c = d >> 5, j = d & 31;   // 128 float4 == one row
    float4 x = *(const float4*)(codebook + (size_t)k * DDIM + d);
    ushort4 h, l;
    cvt_hilo4(x, h, l);
    unsigned short* p = cbPk + (size_t)k * (CBROWB / 2) + c * 64 + j;
    *(ushort4*)p = h;
    *(ushort4*)(p + 32) = l;

    float s = fmaf(x.x, x.x, fmaf(x.y, x.y, fmaf(x.z, x.z, x.w * x.w)));
    #pragma unroll
    for (int off = 16; off >= 1; off >>= 1) s += __shfl_down(s, off, 32);
    __shared__ float red[4];
    if ((tid & 31) == 0) red[tid >> 5] = s;
    __syncthreads();
    if (tid == 0) cbSq[k] = red[0] + red[1] + red[2] + red[3];
}

// ----------------------- fused GEMM + argmin (WMMA) ------------------------
__global__ __launch_bounds__(128) void vq_argmin_kernel(
    const float* __restrict__ codes, const unsigned short* __restrict__ cbPk,
    const float* __restrict__ cbSq, int* __restrict__ ids,
    float* __restrict__ idsF)
{
    __shared__ __align__(16) unsigned char sA[MT * AROWB];        // codes tile, whole D
    __shared__ __align__(16) unsigned char sB[2][NT * BROWB];     // codebook chunk, dbuf

    const int tid  = threadIdx.x;
    const int wv   = tid >> 5;      // wave 0..3 -> 16 code rows each
    const int lane = tid & 31;
    const int half = lane >> 4;     // C rows r / r+8 split
    const int l15  = lane & 15;     // C column within 16-wide tile
    const int m0   = blockIdx.x * MT;

    // ---- stage codes tile ONCE: f32 -> bf16 hi/lo, chunk-interleaved ----
    for (int v = tid; v < MT * DDIM / 4; v += 128) {
        int row = v >> 7, f4 = v & 127;           // 128 float4 per row
        int d = f4 * 4, c = d >> 5, j = d & 31;
        float4 x = *(const float4*)(codes + (size_t)(m0 + row) * DDIM + d);
        ushort4 h, l;
        cvt_hilo4(x, h, l);
        unsigned char* p = sA + row * AROWB + c * 128 + j * 2;
        *(ushort4*)p = h;
        *(ushort4*)(p + 64) = l;
    }

    // async B staging: each thread copies 64B (4 x b128) of the 8KB chunk
    const int bRow = tid >> 1, bPart = tid & 1;
    auto stageB = [&](int gn, int bufn) {
        const int ntn = gn >> 4, cn = gn & (NCH - 1);
        unsigned long long ga = (unsigned long long)(const void*)(
            (const unsigned char*)cbPk + (size_t)(ntn * NT + bRow) * CBROWB +
            cn * 128 + bPart * 64);
        unsigned la = (unsigned)(unsigned long long)(const void*)(
            &sB[bufn][bRow * BROWB + bPart * 64]);
        asm volatile(
            "global_load_async_to_lds_b128 %0, %1, off\n\t"
            "global_load_async_to_lds_b128 %0, %1, off offset:16\n\t"
            "global_load_async_to_lds_b128 %0, %1, off offset:32\n\t"
            "global_load_async_to_lds_b128 %0, %1, off offset:48"
            :: "v"(la), "v"(ga) : "memory");
    };

    float bestV[8];
    int   bestK[8];
    #pragma unroll
    for (int r = 0; r < 8; ++r) { bestV[r] = 3.4e38f; bestK[r] = 0; }

    const unsigned char* aPtr = sA + (wv * 16 + l15) * AROWB + half * 32;

    stageB(0, 0);
    asm volatile("s_wait_asynccnt 0x0" ::: "memory");
    __syncthreads();   // also covers the sA ds-stores above

    for (int nt = 0; nt < NTILES; ++nt) {
        // zero accumulators once per N-tile (structural, not predicated)
        v8f acc[4];
        acc[0] = (v8f){}; acc[1] = (v8f){}; acc[2] = (v8f){}; acc[3] = (v8f){};

        #pragma unroll 2
        for (int c = 0; c < NCH; ++c) {          // buf parity == c & 1
            const int gn = nt * NCH + c + 1;
            if (gn < GTOT) stageB(gn, (c + 1) & 1);   // overlap copy with math

            const unsigned char* ap = aPtr + c * 128;
            v16bf aHi = *(const v16bf*)ap;
            v16bf aLo = *(const v16bf*)(ap + 64);
            #pragma unroll
            for (int t = 0; t < 4; ++t) {
                const unsigned char* bp = &sB[c & 1][(t * 16 + l15) * BROWB + half * 32];
                v16bf bHi = *(const v16bf*)bp;
                v16bf bLo = *(const v16bf*)(bp + 64);
                v8f a = acc[t];
                a = __builtin_amdgcn_wmma_f32_16x16x32_bf16(false, aHi, false, bHi, (short)0, a, false, false);
                a = __builtin_amdgcn_wmma_f32_16x16x32_bf16(false, aHi, false, bLo, (short)0, a, false, false);
                a = __builtin_amdgcn_wmma_f32_16x16x32_bf16(false, aLo, false, bHi, (short)0, a, false, false);
                acc[t] = a;
            }
            asm volatile("s_wait_asynccnt 0x0" ::: "memory");
            __syncthreads();
        }

        // score = ||c_k||^2 - 2*cross ; k ascending => strict '<' keeps first
        const int n0 = nt * NT;
        #pragma unroll
        for (int t = 0; t < 4; ++t) {
            int k = n0 + t * 16 + l15;
            float sk = cbSq[k];
            #pragma unroll
            for (int r = 0; r < 8; ++r) {
                float sc = fmaf(-2.0f, acc[t][r], sk);
                if (sc < bestV[r]) { bestV[r] = sc; bestK[r] = k; }
            }
        }
    }

    // argmin across the 16 lanes holding different columns (stays in half)
    #pragma unroll
    for (int off = 8; off >= 1; off >>= 1) {
        #pragma unroll
        for (int r = 0; r < 8; ++r) {
            float ov = __shfl_xor(bestV[r], off, 32);
            int   ok = __shfl_xor(bestK[r], off, 32);
            if (ov < bestV[r] || (ov == bestV[r] && ok < bestK[r])) {
                bestV[r] = ov; bestK[r] = ok;
            }
        }
    }

    if (l15 == 0) {
        int mBase = m0 + wv * 16 + half * 8;   // C row = r + 8*half
        #pragma unroll
        for (int r = 0; r < 8; ++r) {
            ids[mBase + r]  = bestK[r];
            idsF[mBase + r] = (float)bestK[r];
        }
    }
}

// --------------------------- gather codebook[id] ---------------------------
__global__ __launch_bounds__(128) void vq_gather_kernel(
    const float* __restrict__ codebook, const int* __restrict__ ids,
    float* __restrict__ outCodes)
{
    const int m  = blockIdx.x;
    const int id = ids[m];
    const float4* src = (const float4*)(codebook + (size_t)id * DDIM);
    float4*       dst = (float4*)(outCodes + (size_t)m * DDIM);
    dst[threadIdx.x] = src[threadIdx.x];     // 128 threads x 16B = 512 floats
}

extern "C" void kernel_launch(void* const* d_in, const int* in_sizes, int n_in,
                              void* d_out, int out_size, void* d_ws, size_t ws_size,
                              hipStream_t stream) {
    const float* codebook = (const float*)d_in[0];   // (4096, 512) f32
    const float* codes    = (const float*)d_in[1];   // (8, 2048, 512) f32

    float* outCodes = (float*)d_out;                           // 16384*512 floats
    float* outIdsF  = outCodes + (size_t)MROWS * DDIM;         // 16384 floats

    unsigned short* cbPk = (unsigned short*)d_ws;                          // 8 MB
    float* cbSq = (float*)((char*)d_ws + (size_t)KCB * CBROWB);            // 16 KB
    int*   ids  = (int*)((char*)cbSq + (size_t)KCB * sizeof(float));       // 64 KB

    vq_prep_kernel<<<KCB, 128, 0, stream>>>(codebook, cbPk, cbSq);
    vq_argmin_kernel<<<MROWS / MT, 128, 0, stream>>>(codes, cbPk, cbSq, ids, outIdsF);
    vq_gather_kernel<<<MROWS, 128, 0, stream>>>(codebook, ids, outCodes);
}
